// WellAwareTransition_88347477279466
// MI455X (gfx1250) — compile-verified, matrix operand order checked by reading, not run on hardware
//
#include <hip/hip_runtime.h>

// ---------------------------------------------------------------------------
// WellAwareTransition fused CDNA5 (gfx1250) implementation.
//
// zt1 = einsum('bk,kij,bj->bi', hz, At_w, zt) + einsum('bk,kiu,bu->bi', hz, Bt_w, ut*dt) + biases
// rewritten as ONE GEMM:  A[b, K'] @ Wfused[K', i]
//   A[b,(k,j)] = hz[b,k]*zt[b,j]     (generated on the fly, never stored)
//   A[b,(k,u)] = hz[b,k]*ut_dt[b,u]
//   bias rows: pseudo-k with hz == 1
// Wfused is repacked once per call into bf16 WMMA-B tile layout.
// Per-well heads are WMMA GEMMs (K padded to 32-chunks), ReLU staged via LDS.
// ---------------------------------------------------------------------------

typedef __bf16 bf16_t;
typedef __attribute__((ext_vector_type(16))) __bf16 v16bf;
typedef __attribute__((ext_vector_type(8)))  float  v8f;

#define WMMA_BF16(a, b, c) \
  __builtin_amdgcn_wmma_f32_16x16x32_bf16(false, (a), false, (b), (short)0, (c), false, false)

__device__ __forceinline__ int kpos(int lane, int t) {
  // documented 16-bit A-matrix 16x32 K placement (mirrored for B):
  // lanes 0-15: K = {0..7, 16..23}; lanes 16-31: K = {8..15, 24..31}
  return (t < 8 ? t : t + 8) + ((lane >> 4) << 3);
}

__device__ __forceinline__ v8f vzero8() {
  v8f v;
#pragma unroll
  for (int i = 0; i < 8; ++i) v[i] = 0.0f;
  return v;
}

// ---------------- problem constants ----------------
constexpr int Bn = 4096, Ln = 128, Un = 25, Pn = 20, NIn = 5, FCn = 128, WINn = 281;
constexpr int KCH_A   = 516;   // 129 k-values * 4 j-chunks (128 j's) for At (+bias row k=128)
constexpr int KCH_TOT = 645;   // + 129 k-values * 1 u-chunk (25->32 padded) for Bt (+bias)
constexpr int KSPLIT  = 8;

// workspace byte offsets
constexpr size_t OFF_HZ   = 0;                       // B*128 f32
constexpr size_t OFF_H1   = OFF_HZ   + (size_t)Bn * 128 * 4;  // B*64 f32
constexpr size_t OFF_UTDT = OFF_H1   + (size_t)Bn * 64 * 4;   // B*32 f32 (padded)
constexpr size_t OFF_ZT1  = OFF_UTDT + (size_t)Bn * 32 * 4;   // B*128 f32 accumulator
constexpr size_t OFF_WF   = OFF_ZT1  + (size_t)Bn * 128 * 4;  // 645*4096 bf16
constexpr size_t OFF_WH1  = OFF_WF   + (size_t)KCH_TOT * 4096 * 2; // 20*9*4*512 bf16
constexpr size_t OFF_WH2  = OFF_WH1  + (size_t)Pn * 9 * 4 * 512 * 2; // 20*2*4*512 bf16

// ---------------------------------------------------------------------------
// Pack kernels: f32 weights -> bf16 WMMA-B tile layout.
// Tile (chunk c, n-block nb): 1024B; lane L owns 32B = 16 bf16 at (n = nb*16+L%16, K = kpos).
// ---------------------------------------------------------------------------
__global__ void k_pack_wfused(const float* __restrict__ At_w, const float* __restrict__ At_b,
                              const float* __restrict__ Bt_w, const float* __restrict__ Bt_b,
                              bf16_t* __restrict__ out) {
  int idx = blockIdx.x * 256 + threadIdx.x;
  if (idx >= KCH_TOT * 8 * 32 * 16) return;
  int t = idx & 15, lane = (idx >> 4) & 31, nb = (idx >> 9) & 7, c = idx >> 12;
  int n = nb * 16 + (lane & 15);
  int K = c * 32 + kpos(lane, t);
  float v;
  if (K < 16512) {                       // At segment: K = k*128 + j, k in [0,129)
    int k = K >> 7, j = K & 127;
    v = (k < 128) ? At_w[(size_t)k * 16384 + n * 128 + j] : At_b[n * 128 + j];
  } else {                               // Bt segment: K = k*32 + u (u padded 25->32)
    int rel = K - 16512;
    int k = rel >> 5, u = rel & 31;
    v = (u < Un) ? ((k < 128) ? Bt_w[(size_t)k * 3200 + n * Un + u] : Bt_b[n * Un + u]) : 0.0f;
  }
  out[(size_t)(c * 8 + nb) * 512 + lane * 16 + t] = (bf16_t)v;
}

__global__ void k_pack_wh1(const float* __restrict__ w1, bf16_t* __restrict__ out) {
  int idx = blockIdx.x * 256 + threadIdx.x;
  if (idx >= Pn * 9 * 4 * 512) return;
  int p = idx / 18432;  int r0 = idx % 18432;
  int c = r0 / 2048;    int r1 = r0 % 2048;
  int nb = r1 >> 9, lane = (r1 >> 4) & 31, t = r1 & 15;
  int n = nb * 16 + (lane & 15);
  int K = c * 32 + kpos(lane, t);
  // head_in row mapping with padding: [zt1:0..127][ut:128..152 pad to 159][wf:160..287]
  int r;
  if (K < 128)      r = K;
  else if (K < 160) { int u = K - 128; r = (u < Un) ? 128 + u : -1; }
  else              r = 153 + (K - 160);
  float v = (r >= 0) ? w1[((size_t)p * WINn + r) * 64 + n] : 0.0f;
  out[(size_t)((p * 9 + c) * 4 + nb) * 512 + lane * 16 + t] = (bf16_t)v;
}

__global__ void k_pack_wh2(const float* __restrict__ w2, bf16_t* __restrict__ out) {
  int idx = blockIdx.x * 256 + threadIdx.x;
  if (idx >= Pn * 2 * 4 * 512) return;
  int p = idx >> 12;  int r0 = idx & 4095;
  int c = r0 >> 11, nb = (r0 >> 9) & 3, lane = (r0 >> 4) & 31, t = r0 & 15;
  int n = nb * 16 + (lane & 15);
  int r = c * 32 + kpos(lane, t);          // K = 64 exactly
  float v = w2[((size_t)p * 64 + r) * 64 + n];
  out[(size_t)((p * 2 + c) * 4 + nb) * 512 + lane * 16 + t] = (bf16_t)v;
}

// ---------------------------------------------------------------------------
// Encoder MLP (tiny, VALU).
// ---------------------------------------------------------------------------
__global__ void k_te1(const float* __restrict__ zt, const float* __restrict__ dt,
                      const float* __restrict__ w1, const float* __restrict__ b1,
                      float* __restrict__ h1) {
  int idx = blockIdx.x * 256 + threadIdx.x;       // B*64
  int b = idx >> 6, o = idx & 63;
  float s = b1[o];
  for (int k = 0; k < 128; ++k) s += zt[b * 128 + k] * w1[k * 64 + o];
  s += dt[b] * w1[128 * 64 + o];
  h1[idx] = fmaxf(s, 0.0f);
}

__global__ void k_te2(const float* __restrict__ h1, const float* __restrict__ w2,
                      const float* __restrict__ b2, const float* __restrict__ ut,
                      const float* __restrict__ dt, float* __restrict__ hz,
                      float* __restrict__ zt1acc, float* __restrict__ utdt) {
  int idx = blockIdx.x * 256 + threadIdx.x;       // B*128
  int b = idx >> 7, o = idx & 127;
  float s = b2[o];
  for (int k = 0; k < 64; ++k) s += h1[b * 64 + k] * w2[k * 128 + o];
  hz[idx] = s;
  zt1acc[idx] = 0.0f;                             // pre-zero accumulator for K2 atomics
  if (o < 32) utdt[b * 32 + o] = (o < Un) ? ut[b * Un + o] * dt[b] : 0.0f;
}

// ---------------------------------------------------------------------------
// Mega-GEMM: zt1[b, 0..127] via bf16 WMMA, K split 8 ways, f32 atomic reduce.
// grid: 256 blocks x 256 threads; 8 waves/block share a K-split (B tiles hit WGP$).
// ---------------------------------------------------------------------------
__global__ void __launch_bounds__(256) k_megagemm(const float* __restrict__ zt,
                                                  const float* __restrict__ hz,
                                                  const float* __restrict__ utdt,
                                                  const bf16_t* __restrict__ wf,
                                                  float* __restrict__ zt1acc) {
  int lane = threadIdx.x & 31;
  int wave = threadIdx.x >> 5;
  int mgroup = blockIdx.x & 31;
  int ks = blockIdx.x >> 5;
  int b0 = (mgroup * 8 + wave) * 16;
  int bb = b0 + (lane & 15);
  int hzbase = bb * 128;

  // per-batch vectors preloaded into registers in WMMA A layout
  float ztA[4][16];
  float udA[16];
#pragma unroll
  for (int jc = 0; jc < 4; ++jc)
#pragma unroll
    for (int t = 0; t < 16; ++t)
      ztA[jc][t] = zt[bb * 128 + jc * 32 + kpos(lane, t)];
#pragma unroll
  for (int t = 0; t < 16; ++t) udA[t] = utdt[bb * 32 + kpos(lane, t)];

  v8f acc[8];
#pragma unroll
  for (int nb = 0; nb < 8; ++nb) acc[nb] = vzero8();

  const v16bf* wfv = (const v16bf*)wf;
  int ka0 = (129 * ks) / KSPLIT, ka1 = (129 * (ks + 1)) / KSPLIT;

  // ---- segment A: At_w (+At_b as k==128) ----
  for (int k = ka0; k < ka1; ++k) {
    float hzk = (k < 128) ? hz[hzbase + k] : 1.0f;
#pragma unroll
    for (int jc = 0; jc < 4; ++jc) {
      v16bf a;
#pragma unroll
      for (int t = 0; t < 16; ++t) a[t] = (bf16_t)(hzk * ztA[jc][t]);
      int c = k * 4 + jc;
#pragma unroll
      for (int nb = 0; nb < 8; ++nb) {
        v16bf bt = wfv[(size_t)(c * 8 + nb) * 32 + lane];
        acc[nb] = WMMA_BF16(a, bt, acc[nb]);
      }
    }
  }
  // ---- segment B: Bt_w (+Bt_b as k==128) ----
  for (int k = ka0; k < ka1; ++k) {
    float hzk = (k < 128) ? hz[hzbase + k] : 1.0f;
    v16bf a;
#pragma unroll
    for (int t = 0; t < 16; ++t) a[t] = (bf16_t)(hzk * udA[t]);
    int c = KCH_A + k;
#pragma unroll
    for (int nb = 0; nb < 8; ++nb) {
      v16bf bt = wfv[(size_t)(c * 8 + nb) * 32 + lane];
      acc[nb] = WMMA_BF16(a, bt, acc[nb]);
    }
  }

  // epilogue: C layout (VGPR r -> M=r(+8), lane%16 -> N), atomic f32 reduce across K-splits
  int hi = lane >> 4, n16 = lane & 15;
#pragma unroll
  for (int nb = 0; nb < 8; ++nb)
#pragma unroll
    for (int r = 0; r < 8; ++r) {
      int m = r + hi * 8;
      unsafeAtomicAdd(&zt1acc[(size_t)(b0 + m) * 128 + nb * 16 + n16], acc[nb][r]);
    }
}

// ---------------------------------------------------------------------------
// Per-well heads: (16b x 288) @ (288 x 64) -> ReLU -> (16 x 64)@(64 x 64) -> ReLU -> 64x2
// ---------------------------------------------------------------------------
__global__ void __launch_bounds__(32) k_heads(const float* __restrict__ zt1acc,
                                              const float* __restrict__ ut,
                                              const float* __restrict__ wfeat,
                                              const bf16_t* __restrict__ wh1p,
                                              const bf16_t* __restrict__ wh2p,
                                              const float* __restrict__ b1,
                                              const float* __restrict__ b2,
                                              const float* __restrict__ w3,
                                              const float* __restrict__ b3,
                                              float* __restrict__ out) {
  __shared__ float lds1[16][64];
  __shared__ float lds2[16][64];
  int lane = threadIdx.x;
  int p = blockIdx.x % Pn;
  int b0 = (blockIdx.x / Pn) * 16;
  int bb = b0 + (lane & 15);

  const v16bf* w1v = (const v16bf*)wh1p;
  const v16bf* w2v = (const v16bf*)wh2p;

  v8f acc1[4];
#pragma unroll
  for (int nb = 0; nb < 4; ++nb) acc1[nb] = vzero8();

  // chunks 0..3: zt1
#pragma unroll
  for (int c = 0; c < 4; ++c) {
    v16bf a;
#pragma unroll
    for (int t = 0; t < 16; ++t)
      a[t] = (bf16_t)zt1acc[bb * 128 + c * 32 + kpos(lane, t)];
#pragma unroll
    for (int nb = 0; nb < 4; ++nb) {
      v16bf bt = w1v[(size_t)((p * 9 + c) * 4 + nb) * 32 + lane];
      acc1[nb] = WMMA_BF16(a, bt, acc1[nb]);
    }
  }
  // chunk 4: ut (padded 25->32)
  {
    v16bf a;
#pragma unroll
    for (int t = 0; t < 16; ++t) {
      int kk = kpos(lane, t);
      a[t] = (bf16_t)((kk < Un) ? ut[bb * Un + kk] : 0.0f);
    }
#pragma unroll
    for (int nb = 0; nb < 4; ++nb) {
      v16bf bt = w1v[(size_t)((p * 9 + 4) * 4 + nb) * 32 + lane];
      acc1[nb] = WMMA_BF16(a, bt, acc1[nb]);
    }
  }
  // chunks 5..8: well features
#pragma unroll
  for (int c = 5; c < 9; ++c) {
    v16bf a;
#pragma unroll
    for (int t = 0; t < 16; ++t)
      a[t] = (bf16_t)wfeat[(size_t)(bb * Pn + p) * FCn + (c - 5) * 32 + kpos(lane, t)];
#pragma unroll
    for (int nb = 0; nb < 4; ++nb) {
      v16bf bt = w1v[(size_t)((p * 9 + c) * 4 + nb) * 32 + lane];
      acc1[nb] = WMMA_BF16(a, bt, acc1[nb]);
    }
  }

  int hi = lane >> 4, n16 = lane & 15;
#pragma unroll
  for (int nb = 0; nb < 4; ++nb)
#pragma unroll
    for (int r = 0; r < 8; ++r) {
      int m = r + hi * 8, n = nb * 16 + n16;
      lds1[m][n] = fmaxf(acc1[nb][r] + b1[p * 64 + n], 0.0f);
    }
  __syncthreads();

  v8f acc2[4];
#pragma unroll
  for (int nb = 0; nb < 4; ++nb) acc2[nb] = vzero8();
#pragma unroll
  for (int c = 0; c < 2; ++c) {
    v16bf a;
#pragma unroll
    for (int t = 0; t < 16; ++t)
      a[t] = (bf16_t)lds1[lane & 15][c * 32 + kpos(lane, t)];
#pragma unroll
    for (int nb = 0; nb < 4; ++nb) {
      v16bf bt = w2v[(size_t)((p * 2 + c) * 4 + nb) * 32 + lane];
      acc2[nb] = WMMA_BF16(a, bt, acc2[nb]);
    }
  }
#pragma unroll
  for (int nb = 0; nb < 4; ++nb)
#pragma unroll
    for (int r = 0; r < 8; ++r) {
      int m = r + hi * 8, n = nb * 16 + n16;
      lds2[m][n] = fmaxf(acc2[nb][r] + b2[p * 64 + n], 0.0f);
    }
  __syncthreads();

  // layer 3: 64 -> 2 (VALU); lane = (m, o)
  int m = lane & 15, o = lane >> 4;
  float s = b3[p * 2 + o];
  for (int i = 0; i < 64; ++i) s += lds2[m][i] * w3[(p * 64 + i) * 2 + o];
  out[(size_t)Bn * 128 + (size_t)(b0 + m) * 45 + p * 2 + o] = s;
}

// ---------------------------------------------------------------------------
// Tail: emit zt1 output + injection head (zu @ inj_w + inj_b).
// ---------------------------------------------------------------------------
__global__ void k_tail(const float* __restrict__ zt1acc, const float* __restrict__ ut,
                       const float* __restrict__ inj_w, const float* __restrict__ inj_b,
                       float* __restrict__ out) {
  int b = blockIdx.x, t = threadIdx.x;
  out[(size_t)b * 128 + t] = zt1acc[(size_t)b * 128 + t];
  if (t < NIn) {
    float s = inj_b[t];
    for (int k = 0; k < 128; ++k) s += zt1acc[(size_t)b * 128 + k] * inj_w[k * NIn + t];
    for (int u = 0; u < Un; ++u) s += ut[b * Un + u] * inj_w[(128 + u) * NIn + t];
    out[(size_t)Bn * 128 + (size_t)b * 45 + 2 * Pn + t] = s;
  }
}

// ---------------------------------------------------------------------------
extern "C" void kernel_launch(void* const* d_in, const int* in_sizes, int n_in,
                              void* d_out, int out_size, void* d_ws, size_t ws_size,
                              hipStream_t stream) {
  (void)in_sizes; (void)n_in; (void)out_size; (void)ws_size;
  const float* zt    = (const float*)d_in[0];
  const float* dt    = (const float*)d_in[1];
  const float* ut    = (const float*)d_in[2];
  const float* wfeat = (const float*)d_in[3];
  const float* te_w1 = (const float*)d_in[4];
  const float* te_b1 = (const float*)d_in[5];
  const float* te_w2 = (const float*)d_in[6];
  const float* te_b2 = (const float*)d_in[7];
  const float* At_w  = (const float*)d_in[8];
  const float* At_b  = (const float*)d_in[9];
  const float* Bt_w  = (const float*)d_in[10];
  const float* Bt_b  = (const float*)d_in[11];
  const float* wh_w1 = (const float*)d_in[12];
  const float* wh_b1 = (const float*)d_in[13];
  const float* wh_w2 = (const float*)d_in[14];
  const float* wh_b2 = (const float*)d_in[15];
  const float* wh_w3 = (const float*)d_in[16];
  const float* wh_b3 = (const float*)d_in[17];
  const float* inj_w = (const float*)d_in[18];
  const float* inj_b = (const float*)d_in[19];
  float* out = (float*)d_out;

  char* ws = (char*)d_ws;
  float*  hz     = (float*)(ws + OFF_HZ);
  float*  h1     = (float*)(ws + OFF_H1);
  float*  utdt   = (float*)(ws + OFF_UTDT);
  float*  zt1acc = (float*)(ws + OFF_ZT1);
  bf16_t* wfp    = (bf16_t*)(ws + OFF_WF);
  bf16_t* wh1p   = (bf16_t*)(ws + OFF_WH1);
  bf16_t* wh2p   = (bf16_t*)(ws + OFF_WH2);

  // weight repack (one-time per call; ~6 MB writes, bandwidth trivial)
  k_pack_wfused<<<(KCH_TOT * 4096 + 255) / 256, 256, 0, stream>>>(At_w, At_b, Bt_w, Bt_b, wfp);
  k_pack_wh1<<<(Pn * 9 * 4 * 512 + 255) / 256, 256, 0, stream>>>(wh_w1, wh1p);
  k_pack_wh2<<<(Pn * 2 * 4 * 512 + 255) / 256, 256, 0, stream>>>(wh_w2, wh2p);

  // encoder MLP + ut*dt + accumulator zeroing
  k_te1<<<(Bn * 64) / 256, 256, 0, stream>>>(zt, dt, te_w1, te_b1, h1);
  k_te2<<<(Bn * 128) / 256, 256, 0, stream>>>(h1, te_w2, te_b2, ut, dt, hz, zt1acc, utdt);

  // fused bilinear mega-GEMM (dominant 20.6 GFLOP via v_wmma_f32_16x16x32_bf16)
  k_megagemm<<<32 * KSPLIT, 256, 0, stream>>>(zt, hz, utdt, wfp, zt1acc);

  // per-well heads (WMMA) + outputs
  k_heads<<<(Bn / 16) * Pn, 32, 0, stream>>>(zt1acc, ut, wfeat, wh1p, wh2p,
                                             wh_b1, wh_b2, wh_w3, wh_b3, out);
  k_tail<<<Bn, 128, 0, stream>>>(zt1acc, ut, inj_w, inj_b, out);
}
